// STM_41575283425930
// MI455X (gfx1250) — matrix operand order repaired and checked by reading, not compiled, not run on hardware
//
#include <hip/hip_runtime.h>

typedef _Float16 f16;
typedef __attribute__((ext_vector_type(16))) _Float16 v16h;
typedef __attribute__((ext_vector_type(8)))  _Float16 v8h;
typedef __attribute__((ext_vector_type(8)))  float    v8f;

#define B_    2
#define NF    5
#define C_    256
#define HW    4096            // 64*64
#define CK    32
#define CV    128
#define MFR   4               // memory frames
#define MKEYS (MFR * HW)      // 16384
#define NORM_EPS 1e-12f
#define BN_EPS   1e-5f

#define WMMA_F16(A, B, Cacc) \
  __builtin_amdgcn_wmma_f32_16x16x32_f16(false, (A), false, (B), (short)0, (Cacc), false, false)

// 16-lane butterfly max via DPP (quad_perm, half-mirror, mirror) — no LDS.
__device__ __forceinline__ float dpp_max16(float x) {
  int t;
  t = __builtin_amdgcn_update_dpp(0, __float_as_int(x), 0xB1, 0xF, 0xF, true);   // xor1
  x = fmaxf(x, __int_as_float(t));
  t = __builtin_amdgcn_update_dpp(0, __float_as_int(x), 0x4E, 0xF, 0xF, true);   // xor2
  x = fmaxf(x, __int_as_float(t));
  t = __builtin_amdgcn_update_dpp(0, __float_as_int(x), 0x141, 0xF, 0xF, true);  // row_half_mirror
  x = fmaxf(x, __int_as_float(t));
  t = __builtin_amdgcn_update_dpp(0, __float_as_int(x), 0x140, 0xF, 0xF, true);  // row_mirror
  x = fmaxf(x, __int_as_float(t));
  return x;
}

// WMMA A-operand fragment (16x32 f16): lane holds one row;
// lanes<16 take K {0..15}, lanes>=16 take K {16..31}.
__device__ __forceinline__ v16h load_a_frag(const f16* row, int half) {
  v8h p0 = *(const v8h*)(row + half * 16);
  v8h p1 = *(const v8h*)(row + half * 16 + 8);
  v16h a;
  #pragma unroll
  for (int i = 0; i < 8; i++) { a[i] = p0[i]; a[8 + i] = p1[i]; }
  return a;
}

// gfx1250 async global -> LDS copy, 16B per lane, tracked by ASYNCcnt.
__device__ __forceinline__ void async_b128(unsigned lds_off, const void* gsrc) {
  asm volatile("global_load_async_to_lds_b128 %0, %1, off"
               :: "v"(lds_off), "v"(gsrc) : "memory");
}

// ---------------------------------------------------------------------------
// f32 -> f16 weight conversion
// ---------------------------------------------------------------------------
__global__ __launch_bounds__(256) void cvt_kernel(
    const float* __restrict__ in, f16* __restrict__ out, int n)
{
  int i = blockIdx.x * 256 + threadIdx.x;
  if (i < n) out[i] = (f16)in[i];
}

// ---------------------------------------------------------------------------
// x [b*nf][C][HW] f32  ->  xT [b*nf][HW][C] f16   (LDS tile transpose)
// ---------------------------------------------------------------------------
__global__ __launch_bounds__(256) void xpose_kernel(
    const float* __restrict__ x, f16* __restrict__ xT)
{
  __shared__ f16 t[32][33];
  int z  = blockIdx.z;                 // b*NF + f
  int p0 = blockIdx.x * 32;
  int c0 = blockIdx.y * 32;
  int tx = threadIdx.x & 31;
  int ty = threadIdx.x >> 5;           // 0..7
  const float* xp = x + ((size_t)z * C_ + c0) * HW + p0;
  #pragma unroll
  for (int i = 0; i < 4; i++)
    t[ty + 8 * i][tx] = (f16)xp[(size_t)(ty + 8 * i) * HW + tx];
  __syncthreads();
  f16* op = xT + ((size_t)z * HW + p0) * C_ + c0;
  #pragma unroll
  for (int i = 0; i < 4; i++)
    op[(size_t)(ty + 8 * i) * C_ + tx] = t[tx][ty + 8 * i];
}

// ---------------------------------------------------------------------------
// Key projection GEMM (O=32) + per-position L2 norm.  out f16 [b][npos][32]
// ---------------------------------------------------------------------------
__global__ __launch_bounds__(128) void key_gemm_kernel(
    const f16* __restrict__ xT, const f16* __restrict__ wh,  // wh [32][C]
    f16* __restrict__ out, int frame0, int posshift)
{
  const int lane = threadIdx.x & 31;
  const int wave = threadIdx.x >> 5;
  const int half = lane >> 4;
  const int l16  = lane & 15;
  int tile = blockIdx.x * 4 + wave;
  int bb   = tile >> posshift;
  int pos  = (tile & ((1 << posshift) - 1)) * 16;
  int npos = 16 << posshift;

  const f16* xrow = xT + ((size_t)(bb * NF + frame0) * HW + pos + l16) * C_;

  v8f acc0 = {}, acc1 = {};
  #pragma unroll
  for (int c0 = 0; c0 < C_; c0 += 32) {
    v16h bx = *(const v16h*)(xrow + c0 + half * 16);
    v16h a0 = load_a_frag(wh + (size_t)l16 * C_ + c0, half);
    v16h a1 = load_a_frag(wh + (size_t)(16 + l16) * C_ + c0, half);
    acc0 = WMMA_F16(a0, bx, acc0);
    acc1 = WMMA_F16(a1, bx, acc1);
  }
  float s = 0.f;
  #pragma unroll
  for (int r = 0; r < 8; r++) s += acc0[r] * acc0[r] + acc1[r] * acc1[r];
  s += __shfl_xor(s, 16, 32);
  float inv = 1.0f / fmaxf(sqrtf(s), NORM_EPS);

  f16* op = out + ((size_t)bb * npos + pos + l16) * CK;
  v8h pk0, pk1;
  #pragma unroll
  for (int r = 0; r < 8; r++) {
    pk0[r] = (f16)(acc0[r] * inv);
    pk1[r] = (f16)(acc1[r] * inv);
  }
  *(v8h*)(op + half * 8)      = pk0;
  *(v8h*)(op + 16 + half * 8) = pk1;
}

// ---------------------------------------------------------------------------
// Value projection GEMM (O=128).  TRANSPOSED_OUT: f16 [b][CV][npos] (mv)
// else f16 [b][npos][CV] (qv).
// ---------------------------------------------------------------------------
template <bool TRANSPOSED_OUT>
__global__ __launch_bounds__(128) void value_gemm_kernel(
    const f16* __restrict__ xT, const f16* __restrict__ wh,  // wh [128][C]
    f16* __restrict__ out, int frame0, int posshift)
{
  const int lane = threadIdx.x & 31;
  const int wave = threadIdx.x >> 5;
  const int half = lane >> 4;
  const int l16  = lane & 15;
  int tile = blockIdx.x * 4 + wave;
  int bb   = tile >> posshift;
  int pos  = (tile & ((1 << posshift) - 1)) * 16;
  int npos = 16 << posshift;

  const f16* xrow = xT + ((size_t)(bb * NF + frame0) * HW + pos + l16) * C_;

  v8f acc[8] = {};
  #pragma unroll
  for (int c0 = 0; c0 < C_; c0 += 32) {
    v16h bx = *(const v16h*)(xrow + c0 + half * 16);
    #pragma unroll
    for (int ot = 0; ot < 8; ot++) {
      v16h a = load_a_frag(wh + (size_t)(ot * 16 + l16) * C_ + c0, half);
      acc[ot] = WMMA_F16(a, bx, acc[ot]);
    }
  }
  int k = pos + l16;
  #pragma unroll
  for (int ot = 0; ot < 8; ot++) {
    if (TRANSPOSED_OUT) {
      #pragma unroll
      for (int r = 0; r < 8; r++) {
        int o = ot * 16 + r + half * 8;
        out[((size_t)bb * CV + o) * npos + k] = (f16)acc[ot][r];
      }
    } else {
      v8h pk;
      #pragma unroll
      for (int r = 0; r < 8; r++) pk[r] = (f16)acc[ot][r];
      *(v8h*)(out + ((size_t)bb * npos + k) * CV + ot * 16 + half * 8) = pk;
    }
  }
}

// ---------------------------------------------------------------------------
// Flash attention v3.
// Block = 256 threads = 8 waves = 4 q-tiles x 2 key segments (split-K).
// K/V chunks (64 keys) stream into LDS via global_load_async_to_lds_b128,
// double-buffered per segment; all 4 waves of a segment share the stream.
// Dynamic LDS layout (bytes):
//   [seg s][buf b]: base = s*40960 + b*20480 : mk 4096 | mv 16384
//   pstage: 81920 + wave*2048  (16x64 f16 P re-layout)
//   merge (after main loop, reuses K/V buffer area): mmax@0, msum@512, macc@1024
// ---------------------------------------------------------------------------
__global__ __launch_bounds__(256) void flash_kernel(
    const f16* __restrict__ qk, const f16* __restrict__ mk,
    const f16* __restrict__ mvT, f16* __restrict__ aggh)
{
  extern __shared__ char smem[];

  const int lane = threadIdx.x & 31;
  const int wave = threadIdx.x >> 5;     // 0..7
  const int qt   = wave & 3;             // q-tile within block
  const int ks   = wave >> 2;            // key segment 0/1
  const int half = lane >> 4;
  const int l16  = lane & 15;
  const int gt   = qt * 32 + lane;       // 0..127 within segment group

  const int blk = blockIdx.x;            // 0..31
  const int bb  = blk >> 6;              // 64 blocks per batch
  const int q0  = ((blk & 63) * 4 + qt) * 16;

  const unsigned lds_base = (unsigned)(size_t)(void*)smem;  // LDS byte offset
  const unsigned segbase  = lds_base + (unsigned)ks * 40960u;

  f16* pbuf = (f16*)(smem + 81920 + wave * 2048);

  v16h a_q = load_a_frag(qk + ((size_t)bb * HW + q0 + l16) * CK, half);

  v16h ones;
  #pragma unroll
  for (int i = 0; i < 16; i++) ones[i] = (f16)1.0f;

  v8f acc[8] = {};
  float rmax[8], rsum[8];
  #pragma unroll
  for (int r = 0; r < 8; r++) { rmax[r] = -3.0e38f; rsum[r] = 0.f; }

  const f16* mkb = mk  + (size_t)bb * MKEYS * CK;
  const f16* mvb = mvT + (size_t)bb * CV * MKEYS;

  const int kseg0  = ks * (MKEYS / 2);
  const int nchunk = (MKEYS / 2) / 64;   // 128

  // issue async loads for one 64-key chunk (10 instr/wave: 2 mk + 8 mv)
  auto issue = [&](int k0, int buf) {
    unsigned dst = segbase + (unsigned)buf * 20480u;
    const char* ksrc = (const char*)(mkb + (size_t)k0 * CK);
    async_b128(dst + (unsigned)gt * 16u,          ksrc + gt * 16);
    async_b128(dst + 2048u + (unsigned)gt * 16u,  ksrc + 2048 + gt * 16);
    const char* vsrc = (const char*)(mvb + (size_t)gt * MKEYS + k0);
    unsigned vdst = dst + 4096u + (unsigned)gt * 128u;
    #pragma unroll
    for (int j = 0; j < 8; j++)
      async_b128(vdst + (unsigned)j * 16u, vsrc + j * 16);
  };

  issue(kseg0, 0);

  for (int c = 0; c < nchunk; c++) {
    int buf = c & 1;
    if (c + 1 < nchunk) {
      issue(kseg0 + (c + 1) * 64, (c + 1) & 1);
      asm volatile("s_wait_asynccnt 10" ::: "memory");   // chunk c landed
    } else {
      asm volatile("s_wait_asynccnt 0" ::: "memory");
    }
    __syncthreads();                                      // all waves: data ready

    const f16* kb = (const f16*)(smem + ks * 40960 + buf * 20480);
    const f16* vb = kb + 2048;                            // +4096 bytes

    // ---- scores for 64 keys: 4 tiles (from LDS) --------------------------
    v8f s[4];
    v8f zero = {};
    #pragma unroll
    for (int j = 0; j < 4; j++) {
      v16h b = *(const v16h*)(kb + (size_t)(j * 16 + l16) * CK + half * 16);
      s[j] = WMMA_F16(a_q, b, zero);
    }

    // ---- online softmax: DPP row max, rescale, exponentiate --------------
    v8f scv;
    #pragma unroll
    for (int r = 0; r < 8; r++) {
      float m = fmaxf(fmaxf(s[0][r], s[1][r]), fmaxf(s[2][r], s[3][r]));
      m = dpp_max16(m);
      float mn = fmaxf(rmax[r], m);
      scv[r] = __expf(rmax[r] - mn);
      rmax[r] = mn;
      #pragma unroll
      for (int j = 0; j < 4; j++) s[j][r] = __expf(s[j][r] - mn);
    }
    #pragma unroll
    for (int t = 0; t < 8; t++) acc[t] *= scv;            // packed f32 muls

    // ---- stage P (16x64 f16) through LDS into A-operand layout -----------
    #pragma unroll
    for (int r = 0; r < 8; r++) {
      int row = r + half * 8;
      #pragma unroll
      for (int j = 0; j < 4; j++)
        pbuf[row * 64 + j * 16 + l16] = (f16)s[j][r];
    }
    asm volatile("s_wait_dscnt 0" ::: "memory");
    v16h ap0 = load_a_frag(pbuf + l16 * 64, half);
    v16h ap1 = load_a_frag(pbuf + l16 * 64 + 32, half);

    // ---- row sums on the matrix pipe: P x ones ---------------------------
    v8f sums = WMMA_F16(ap0, ones, zero);
    sums = WMMA_F16(ap1, ones, sums);
    #pragma unroll
    for (int r = 0; r < 8; r++) rsum[r] = rsum[r] * scv[r] + sums[r];

    // ---- P·V: 8 cv-tiles x 2 K-halves (V from LDS) -----------------------
    #pragma unroll
    for (int nt = 0; nt < 8; nt++) {
      const f16* vrow = vb + (size_t)(nt * 16 + l16) * 64 + half * 16;
      v16h bv0 = *(const v16h*)(vrow);
      v16h bv1 = *(const v16h*)(vrow + 32);
      acc[nt] = WMMA_F16(ap0, bv0, acc[nt]);
      acc[nt] = WMMA_F16(ap1, bv1, acc[nt]);
    }
    __syncthreads();                     // safe to refill this buffer
  }

  // ---- 2-way split-K merge: wave (qt,ks=1) -> wave (qt,ks=0) -------------
  float* mmax = (float*)smem;            // [8][16]
  float* msum = (float*)(smem + 512);    // [4][16]
  float* macc = (float*)(smem + 1024);   // [4][16][CV]

  if (l16 == 0) {
    #pragma unroll
    for (int r = 0; r < 8; r++) mmax[wave * 16 + r + half * 8] = rmax[r];
  }
  __syncthreads();

  float gmax[8];
  #pragma unroll
  for (int r = 0; r < 8; r++) {
    int row = r + half * 8;
    gmax[r] = fmaxf(mmax[qt * 16 + row], mmax[(qt + 4) * 16 + row]);
  }
  #pragma unroll
  for (int r = 0; r < 8; r++) {
    float fct = __expf(rmax[r] - gmax[r]);
    rsum[r] *= fct;
    #pragma unroll
    for (int t = 0; t < 8; t++) acc[t][r] *= fct;
  }
  if (ks == 1) {
    #pragma unroll
    for (int r = 0; r < 8; r++) {
      int row = r + half * 8;
      if (l16 == 0) msum[qt * 16 + row] = rsum[r];
      #pragma unroll
      for (int t = 0; t < 8; t++)
        macc[(qt * 16 + row) * CV + t * 16 + l16] = acc[t][r];
    }
  }
  __syncthreads();
  if (ks == 0) {
    #pragma unroll
    for (int r = 0; r < 8; r++) {
      int row = r + half * 8;
      float stot = rsum[r] + msum[qt * 16 + row];
      float sinv = 1.0f / fmaxf(stot, 1e-30f);
      #pragma unroll
      for (int t = 0; t < 8; t++) {
        float v = acc[t][r] + macc[(qt * 16 + row) * CV + t * 16 + l16];
        aggh[((size_t)bb * HW + q0 + row) * CV + t * 16 + l16] = (f16)(v * sinv);
      }
    }
  }
}

// ---------------------------------------------------------------------------
// sm GEMM: y = sm_w(256x256) x concat(qv, agg)  + BN + ReLU -> out frame 4
// ---------------------------------------------------------------------------
__global__ __launch_bounds__(128) void sm_gemm_kernel(
    const f16* __restrict__ qvh,  const f16* __restrict__ aggh,
    const f16* __restrict__ smh,  // [256][256] f16
    const float* __restrict__ bn_scale, const float* __restrict__ bn_bias,
    const float* __restrict__ bn_mean,  const float* __restrict__ bn_var,
    float* __restrict__ out)
{
  const int lane = threadIdx.x & 31;
  const int wave = threadIdx.x >> 5;
  const int half = lane >> 4;
  const int l16  = lane & 15;
  int tile = blockIdx.x * 4 + wave;       // 0..255 pos tiles
  int pos  = tile * 16;
  int o0   = blockIdx.y * 128;
  int bb   = blockIdx.z;

  const f16* q = qvh  + ((size_t)bb * HW + pos + l16) * CV;
  const f16* g = aggh + ((size_t)bb * HW + pos + l16) * CV;

  v8f acc[8] = {};
  #pragma unroll
  for (int c0 = 0; c0 < C_; c0 += 32) {
    const f16* src = (c0 < CV) ? (q + c0) : (g + (c0 - CV));
    v16h bx = *(const v16h*)(src + half * 16);
    #pragma unroll
    for (int ot = 0; ot < 8; ot++) {
      v16h a = load_a_frag(smh + (size_t)(o0 + ot * 16 + l16) * C_ + c0, half);
      acc[ot] = WMMA_F16(a, bx, acc[ot]);
    }
  }
  int p = pos + l16;
  #pragma unroll
  for (int ot = 0; ot < 8; ot++) {
    #pragma unroll
    for (int r = 0; r < 8; r++) {
      int o = o0 + ot * 16 + r + half * 8;
      float inv = bn_scale[o] * rsqrtf(bn_var[o] + BN_EPS);
      float y = (acc[ot][r] - bn_mean[o]) * inv + bn_bias[o];
      out[(((size_t)bb * NF + (NF - 1)) * C_ + o) * HW + p] = fmaxf(y, 0.f);
    }
  }
}

// ---------------------------------------------------------------------------
extern "C" void kernel_launch(void* const* d_in, const int* in_sizes, int n_in,
                              void* d_out, int out_size, void* d_ws, size_t ws_size,
                              hipStream_t stream)
{
  (void)in_sizes; (void)n_in; (void)out_size; (void)ws_size;

  const float* x        = (const float*)d_in[0];
  // d_in[1] = prev_mask (unused by reference)
  const float* qk_w     = (const float*)d_in[2];
  const float* qv_w     = (const float*)d_in[3];
  const float* mk_w     = (const float*)d_in[4];
  const float* mv_w     = (const float*)d_in[5];
  const float* sm_w     = (const float*)d_in[6];
  const float* bn_scale = (const float*)d_in[7];
  const float* bn_bias  = (const float*)d_in[8];
  const float* bn_mean  = (const float*)d_in[9];
  const float* bn_var   = (const float*)d_in[10];
  float* out = (float*)d_out;

  char* ws = (char*)d_ws;
  f16* xh    = (f16*)ws; ws += (size_t)B_ * NF * HW * C_ * sizeof(f16);  // 21 MB
  f16* wh_qk = (f16*)ws; ws += (size_t)CK * C_ * sizeof(f16);
  f16* wh_qv = (f16*)ws; ws += (size_t)CV * C_ * sizeof(f16);
  f16* wh_mk = (f16*)ws; ws += (size_t)CK * C_ * sizeof(f16);
  f16* wh_mv = (f16*)ws; ws += (size_t)CV * C_ * sizeof(f16);
  f16* wh_sm = (f16*)ws; ws += (size_t)C_ * C_ * sizeof(f16);
  f16* qk_h  = (f16*)ws; ws += (size_t)B_ * HW * CK * sizeof(f16);       // 0.5 MB
  f16* mk_h  = (f16*)ws; ws += (size_t)B_ * MKEYS * CK * sizeof(f16);    // 2 MB
  f16* mv_h  = (f16*)ws; ws += (size_t)B_ * CV * MKEYS * sizeof(f16);    // 8 MB
  f16* qv_h  = (f16*)ws; ws += (size_t)B_ * HW * CV * sizeof(f16);       // 2 MB
  f16* agg_h = (f16*)ws;                                                 // 2 MB

  // frames 0..3 pass straight through
  for (int bb = 0; bb < B_; bb++) {
    size_t off = (size_t)bb * NF * C_ * HW;
    hipMemcpyAsync(out + off, x + off,
                   (size_t)MFR * C_ * HW * sizeof(float),
                   hipMemcpyDeviceToDevice, stream);
  }

  cvt_kernel<<<(CK * C_ + 255) / 256, 256, 0, stream>>>(qk_w, wh_qk, CK * C_);
  cvt_kernel<<<(CV * C_ + 255) / 256, 256, 0, stream>>>(qv_w, wh_qv, CV * C_);
  cvt_kernel<<<(CK * C_ + 255) / 256, 256, 0, stream>>>(mk_w, wh_mk, CK * C_);
  cvt_kernel<<<(CV * C_ + 255) / 256, 256, 0, stream>>>(mv_w, wh_mv, CV * C_);
  cvt_kernel<<<(C_ * C_ + 255) / 256, 256, 0, stream>>>(sm_w, wh_sm, C_ * C_);

  xpose_kernel<<<dim3(HW / 32, C_ / 32, B_ * NF), 256, 0, stream>>>(x, xh);

  key_gemm_kernel<<<(B_ * (HW / 16)) / 4, 128, 0, stream>>>(xh, wh_qk, qk_h, NF - 1, 8);
  key_gemm_kernel<<<(B_ * (MKEYS / 16)) / 4, 128, 0, stream>>>(xh, wh_mk, mk_h, 0, 10);
  value_gemm_kernel<false><<<(B_ * (HW / 16)) / 4, 128, 0, stream>>>(xh, wh_qv, qv_h, NF - 1, 8);
  value_gemm_kernel<true><<<(B_ * (MKEYS / 16)) / 4, 128, 0, stream>>>(xh, wh_mv, mv_h, 0, 10);

  // 32 blocks x 256 threads, 96 KB dynamic LDS
  flash_kernel<<<dim3((B_ * (HW / 16)) / 4 / 4), 256, 98304, stream>>>(
      qk_h, mk_h, mv_h, agg_h);

  sm_gemm_kernel<<<dim3(HW / 16 / 4, 2, B_), 128, 0, stream>>>(
      qv_h, agg_h, wh_sm, bn_scale, bn_bias, bn_mean, bn_var, out);
}
